// RSSM_489626271800
// MI455X (gfx1250) — matrix operand order repaired and checked by reading, not compiled, not run on hardware
//
#include <hip/hip_runtime.h>
#include <hip/hip_bf16.h>
#include <cstdint>

// ---------------- problem dims ----------------
#define B_    128
#define T_    64
#define STOCH 64
#define DET   2048
#define HID   2048
#define ACTD  64
#define OBS   2048
#define KSPLIT 8          // split-K chunks for the skinny head GEMMs

static constexpr size_t BT64 = (size_t)B_ * T_ * STOCH;   // per-head output block

typedef __attribute__((ext_vector_type(16))) __bf16 v16bf;
typedef __attribute__((ext_vector_type(8)))  float  v8f;

union FragU { v16bf v; uint4 q[2]; };

// D/C 16x16 f32: lane holds fixed column (lane&15), rows m0 + r + (lane>=16 ? 8:0).
template<int ACTF, bool WF, bool WB>
__device__ __forceinline__ void store_tile(v8f acc, int m0, int n0, int lane,
                                           const float* __restrict__ bias,
                                           float* __restrict__ Cf,
                                           __bf16* __restrict__ Cb, int N) {
  const int col   = n0 + (lane & 15);
  const int rbase = m0 + ((lane >> 4) << 3);
  const float bv  = bias ? bias[col] : 0.0f;
#pragma unroll
  for (int r = 0; r < 8; ++r) {
    float x = acc[r] + bv;
    if (ACTF) x = x > 0.0f ? x : (__expf(x) - 1.0f);         // ELU
    if (WF) Cf[(size_t)(rbase + r) * N + col] = x;
    if (WB) Cb[(size_t)(rbase + r) * N + col] = (__bf16)x;
  }
}

// ---------------- bf16 WMMA GEMM: C[128 x N] = act(A[128 x K] + bias) -----------
// Block: 128 threads = 4 waves; wave w owns a 32x32 tile (2x2 WMMA tiles) at rows
// w*32. Block tile = 128 x 32; grid.x = N/32. grid.y = K-chunks (split-K): chunk
// c covers k in [c*Kc, c*Kc+Kc) and writes Cf + c*128*N (deterministic partials).
// REQUIRES: Kc and K multiples of 128 (true for all launches below).
// Fragment layouts per CDNA5 ISA 7.12.2:
//   A (16x32, row-major, lda): lane<16 -> K runs {0-7,16-23}; lane>=16 -> {8-15,24-31}
//   B (32x16) from pre-transposed Wt[N x K]: lane<16 -> K {0-7,16-23}; lane>=16 -> {8-15,24-31}
template<int ACTF, bool WF, bool WB>
__global__ __launch_bounds__(128) void gemm_bf16_wmma(
    const __bf16* __restrict__ A, int lda,
    const __bf16* __restrict__ Wt,
    const float* __restrict__ bias,
    float* __restrict__ Cf, __bf16* __restrict__ Cb,
    int N, int K, int Kc)
{
  const int lane = threadIdx.x & 31;
  const int wave = threadIdx.x >> 5;
  const int m0 = wave * 32;
  const int n0 = blockIdx.x * 32;
  const int kbeg = blockIdx.y * Kc;
  const int kend = (kbeg + Kc < K) ? (kbeg + Kc) : K;
  if (WF) Cf += (size_t)blockIdx.y * 128 * N;   // split-K partial slab

  const int lr = lane & 15, lh = lane >> 4;
  // per-lane base pointers, hoisted; loop body uses constant offsets only
  const __bf16* pa0 = A  + (size_t)(m0 + lr) * lda + (lh << 3) + kbeg;
  const __bf16* pa1 = pa0 + (size_t)16 * lda;
  const __bf16* pb0 = Wt + (size_t)(n0 + lr) * K  + (lh << 3) + kbeg;
  const __bf16* pb1 = pb0 + (size_t)16 * K;

  v8f acc00 = {}, acc01 = {}, acc10 = {}, acc11 = {};
  for (int k0 = kbeg; k0 < kend; k0 += 128) {
    // pull next 256B weight slab into the WGP cache (scope 0 = all levels);
    // speculative prefetch silently drops invalid addresses past the end.
    asm volatile("global_prefetch_b8 %0, off" :: "v"(pb0 + 128));
    asm volatile("global_prefetch_b8 %0, off" :: "v"(pb1 + 128));
#pragma unroll
    for (int u = 0; u < 4; ++u) {
      const int o = u * 32;
      FragU a0, a1, b0, b1;
      a0.q[0] = *(const uint4*)(pa0 + o); a0.q[1] = *(const uint4*)(pa0 + o + 16);
      a1.q[0] = *(const uint4*)(pa1 + o); a1.q[1] = *(const uint4*)(pa1 + o + 16);
      b0.q[0] = *(const uint4*)(pb0 + o); b0.q[1] = *(const uint4*)(pb0 + o + 16);
      b1.q[0] = *(const uint4*)(pb1 + o); b1.q[1] = *(const uint4*)(pb1 + o + 16);
      acc00 = __builtin_amdgcn_wmma_f32_16x16x32_bf16(false, a0.v, false, b0.v, (short)0, acc00, false, false);
      acc01 = __builtin_amdgcn_wmma_f32_16x16x32_bf16(false, a0.v, false, b1.v, (short)0, acc01, false, false);
      acc10 = __builtin_amdgcn_wmma_f32_16x16x32_bf16(false, a1.v, false, b0.v, (short)0, acc10, false, false);
      acc11 = __builtin_amdgcn_wmma_f32_16x16x32_bf16(false, a1.v, false, b1.v, (short)0, acc11, false, false);
    }
    pa0 += 128; pa1 += 128; pb0 += 128; pb1 += 128;
  }
  store_tile<ACTF, WF, WB>(acc00, m0,      n0,      lane, bias, Cf, Cb, N);
  store_tile<ACTF, WF, WB>(acc01, m0,      n0 + 16, lane, bias, Cf, Cb, N);
  store_tile<ACTF, WF, WB>(acc10, m0 + 16, n0,      lane, bias, Cf, Cb, N);
  store_tile<ACTF, WF, WB>(acc11, m0 + 16, n0 + 16, lane, bias, Cf, Cb, N);
}

// ---------------- weight convert + transpose: W[K x N] f32 -> Wt[N x K] bf16 ----
__global__ void k_transpose_bf16(const float* __restrict__ W, __bf16* __restrict__ Wt,
                                 int K, int N) {
  size_t idx = (size_t)blockIdx.x * blockDim.x + threadIdx.x;
  if (idx >= (size_t)K * N) return;
  int k = (int)(idx / N), n = (int)(idx % N);
  Wt[(size_t)n * K + k] = (__bf16)W[idx];
}

// ---------------- per-timestep elementwise kernels ------------------------------
__global__ void k_build_xin(const float* __restrict__ actions, const float* __restrict__ stochf,
                            __bf16* __restrict__ x_in, int t) {
  int idx = blockIdx.x * blockDim.x + threadIdx.x;
  if (idx >= B_ * 128) return;
  int b = idx >> 7, j = idx & 127;
  float v = (j < ACTD) ? actions[((size_t)b * T_ + t) * ACTD + j]
                       : stochf[b * STOCH + (j - ACTD)];
  x_in[idx] = (__bf16)v;
}

__global__ void k_gru_combine(const float* __restrict__ gx, const float* __restrict__ gh,
                              const float* __restrict__ gb, const float* __restrict__ obss,
                              float* __restrict__ detf, __bf16* __restrict__ zyin, int t) {
  int idx = blockIdx.x * blockDim.x + threadIdx.x;
  if (idx >= B_ * DET) return;
  int b = idx / DET, j = idx - b * DET;
  size_t r6 = (size_t)b * (3 * DET);
  float zx = gx[r6 + j]           + gh[r6 + j]           + gb[j];
  float rx = gx[r6 + DET + j]     + gh[r6 + DET + j]     + gb[DET + j];
  float z  = 1.0f / (1.0f + __expf(-zx));
  float r  = 1.0f / (1.0f + __expf(-rx));
  float ax = gx[r6 + 2 * DET + j] + r * gh[r6 + 2 * DET + j] + gb[2 * DET + j];
  float a  = tanhf(ax);
  float h  = detf[idx];
  float d  = (1.0f - z) * h + z * a;
  detf[idx] = d;
  size_t zr = (size_t)b * (2 * DET);
  zyin[zr + j]       = (__bf16)d;                                   // det half (next A)
  zyin[zr + DET + j] = (__bf16)obss[((size_t)b * T_ + t) * OBS + j]; // obs half
}

// reduce KSPLIT partials of the prior head, softplus, write mh/sh
__global__ void k_head_h(const float* __restrict__ hm_parts, const float* __restrict__ bias,
                         float* __restrict__ out, int t) {
  int idx = blockIdx.x * blockDim.x + threadIdx.x;
  if (idx >= B_ * STOCH) return;
  int b = idx / STOCH, j = idx - b * STOCH;
  float m = bias[j], sr = bias[STOCH + j];
#pragma unroll
  for (int c = 0; c < KSPLIT; ++c) {
    m  += hm_parts[(size_t)c * B_ * 128 + b * 128 + j];
    sr += hm_parts[(size_t)c * B_ * 128 + b * 128 + STOCH + j];
  }
  float s = (sr > 20.0f ? sr : log1pf(__expf(sr))) + 0.1f;
  size_t o = ((size_t)b * T_ + t) * STOCH + j;
  out[o]        = m;   // mh
  out[BT64 + o] = s;   // sh
}

__device__ __forceinline__ float hash_u01(uint32_t x) {
  x ^= x >> 17; x *= 0xed5ad4bbu; x ^= x >> 11; x *= 0xac4c1b51u;
  x ^= x >> 15; x *= 0x31848babu; x ^= x >> 14;
  return (float)(x >> 8) * (1.0f / 16777216.0f) + 1e-7f;
}

// reduce KSPLIT partials of posterior head, softplus, sample, write mz/sz/feat
__global__ void k_finalize(const float* __restrict__ zm_parts, const float* __restrict__ bias,
                           const float* __restrict__ detf,
                           float* __restrict__ out, float* __restrict__ stochf, int t) {
  int idx = blockIdx.x * blockDim.x + threadIdx.x;
  if (idx >= B_ * (STOCH + DET)) return;
  int b = idx / (STOCH + DET), j = idx - b * (STOCH + DET);
  float* feat = out + 4 * BT64 + ((size_t)b * T_ + t) * (STOCH + DET);
  if (j < STOCH) {
    float m = bias[j], sr = bias[STOCH + j];
#pragma unroll
    for (int c = 0; c < KSPLIT; ++c) {
      m  += zm_parts[(size_t)c * B_ * 128 + b * 128 + j];
      sr += zm_parts[(size_t)c * B_ * 128 + b * 128 + STOCH + j];
    }
    float s = (sr > 20.0f ? sr : log1pf(__expf(sr))) + 0.1f;
    size_t o = ((size_t)b * T_ + t) * STOCH + j;
    out[2 * BT64 + o] = m;   // mz
    out[3 * BT64 + o] = s;   // sz
    uint32_t seed = (uint32_t)(t * 790151u + (uint32_t)b * 2099u + (uint32_t)j * 41u);
    float u1 = hash_u01(seed * 2654435761u + 1u);
    float u2 = hash_u01(seed * 2246822519u + 7u);
    float n  = sqrtf(-2.0f * __logf(u1)) * __cosf(6.2831853f * u2);
    float smp = m + s * n;
    stochf[b * STOCH + j] = smp;   // carry to next step
    feat[j] = smp;
  } else {
    feat[j] = detf[(size_t)b * DET + (j - STOCH)];
  }
}

// ---------------- host driver ----------------------------------------------------
extern "C" void kernel_launch(void* const* d_in, const int* in_sizes, int n_in,
                              void* d_out, int out_size, void* d_ws, size_t ws_size,
                              hipStream_t stream) {
  (void)in_sizes; (void)n_in; (void)out_size; (void)ws_size;
  const float* obss    = (const float*)d_in[0];
  const float* actions = (const float*)d_in[1];
  const float* h1_w    = (const float*)d_in[2];
  const float* h1_b    = (const float*)d_in[3];
  const float* gru_wi  = (const float*)d_in[4];
  const float* gru_wh  = (const float*)d_in[5];
  const float* gru_b   = (const float*)d_in[6];
  const float* h2_w    = (const float*)d_in[7];
  const float* h2_b    = (const float*)d_in[8];
  const float* h3_w    = (const float*)d_in[9];
  const float* h3_b    = (const float*)d_in[10];
  const float* z_h1_w  = (const float*)d_in[11];
  const float* z_h1_b  = (const float*)d_in[12];
  const float* z_h2_w  = (const float*)d_in[13];
  const float* z_h2_b  = (const float*)d_in[14];
  float* out = (float*)d_out;

  char* ws = (char*)d_ws; size_t off = 0;
  auto alloc = [&](size_t bytes) -> void* {
    void* p = ws + off; off += (bytes + 255) & ~(size_t)255; return p;
  };
  // bf16 transposed weights (Wt[N x K])
  __bf16* wt_h1  = (__bf16*)alloc((size_t)128 * DET * 2);
  __bf16* wt_gwi = (__bf16*)alloc((size_t)DET * 3 * DET * 2);
  __bf16* wt_gwh = (__bf16*)alloc((size_t)DET * 3 * DET * 2);
  __bf16* wt_h2  = (__bf16*)alloc((size_t)DET * HID * 2);
  __bf16* wt_h3  = (__bf16*)alloc((size_t)HID * 128 * 2);
  __bf16* wt_zh1 = (__bf16*)alloc((size_t)(DET + OBS) * HID * 2);
  __bf16* wt_zh2 = (__bf16*)alloc((size_t)HID * 128 * 2);
  // activations
  __bf16* x_in   = (__bf16*)alloc((size_t)B_ * 128 * 2);
  __bf16* xbuf   = (__bf16*)alloc((size_t)B_ * DET * 2);
  float*  gx     = (float*) alloc((size_t)B_ * 3 * DET * 4);
  float*  gh     = (float*) alloc((size_t)B_ * 3 * DET * 4);
  __bf16* zyin   = (__bf16*)alloc((size_t)B_ * 2 * DET * 2);  // [det | obs] bf16
  float*  detf   = (float*) alloc((size_t)B_ * DET * 4);
  __bf16* x2     = (__bf16*)alloc((size_t)B_ * HID * 2);
  __bf16* ybuf   = (__bf16*)alloc((size_t)B_ * HID * 2);
  float*  hm     = (float*) alloc((size_t)KSPLIT * B_ * 128 * 4);  // split-K partials
  float*  zm     = (float*) alloc((size_t)KSPLIT * B_ * 128 * 4);
  float*  stochf = (float*) alloc((size_t)B_ * STOCH * 4);

  // ---- one-time (per launch): convert+transpose weights, zero carried state ----
  auto tr = [&](const float* W, __bf16* Wt, int K, int N) {
    size_t tot = (size_t)K * N;
    k_transpose_bf16<<<(unsigned)((tot + 255) / 256), 256, 0, stream>>>(W, Wt, K, N);
  };
  tr(h1_w,   wt_h1,  128,       DET);
  tr(gru_wi, wt_gwi, DET,       3 * DET);
  tr(gru_wh, wt_gwh, DET,       3 * DET);
  tr(h2_w,   wt_h2,  DET,       HID);
  tr(h3_w,   wt_h3,  HID,       128);
  tr(z_h1_w, wt_zh1, DET + OBS, HID);
  tr(z_h2_w, wt_zh2, HID,       128);
  hipMemsetAsync(stochf, 0, (size_t)B_ * STOCH * 4, stream);
  hipMemsetAsync(detf,   0, (size_t)B_ * DET * 4,   stream);
  hipMemsetAsync(zyin,   0, (size_t)B_ * 2 * DET * 2, stream);

  // ---- sequential scan over T timesteps ----
  for (int t = 0; t < T_; ++t) {
    k_build_xin<<<(B_ * 128 + 255) / 256, 256, 0, stream>>>(actions, stochf, x_in, t);
    // x = elu([act|stoch] @ h1_w + b)  -> bf16
    gemm_bf16_wmma<1, false, true><<<dim3(DET / 32, 1), 128, 0, stream>>>(
        x_in, 128, wt_h1, h1_b, nullptr, xbuf, DET, 128, 128);
    // gx = x @ gru_wi
    gemm_bf16_wmma<0, true, false><<<dim3((3 * DET) / 32, 1), 128, 0, stream>>>(
        xbuf, DET, wt_gwi, nullptr, gx, nullptr, 3 * DET, DET, DET);
    // gh = h_prev @ gru_wh  (h_prev = det half of zyin)
    gemm_bf16_wmma<0, true, false><<<dim3((3 * DET) / 32, 1), 128, 0, stream>>>(
        zyin, 2 * DET, wt_gwh, nullptr, gh, nullptr, 3 * DET, DET, DET);
    // GRU gates -> new det; also stage [det|obs_t] bf16
    k_gru_combine<<<(B_ * DET + 255) / 256, 256, 0, stream>>>(
        gx, gh, gru_b, obss, detf, zyin, t);
    // x2 = elu(det @ h2_w + b)
    gemm_bf16_wmma<1, false, true><<<dim3(HID / 32, 1), 128, 0, stream>>>(
        zyin, 2 * DET, wt_h2, h2_b, nullptr, x2, HID, DET, DET);
    // prior head: x2 @ h3_w (split-K partials; bias folded into reduction)
    gemm_bf16_wmma<0, true, false><<<dim3(128 / 32, KSPLIT), 128, 0, stream>>>(
        x2, HID, wt_h3, nullptr, hm, nullptr, 128, HID, HID / KSPLIT);
    k_head_h<<<(B_ * STOCH + 255) / 256, 256, 0, stream>>>(hm, h3_b, out, t);
    // y = elu([det|obs] @ z_h1_w + b)
    gemm_bf16_wmma<1, false, true><<<dim3(HID / 32, 1), 128, 0, stream>>>(
        zyin, 2 * DET, wt_zh1, z_h1_b, nullptr, ybuf, HID, 2 * DET, 2 * DET);
    // posterior head: y @ z_h2_w (split-K partials)
    gemm_bf16_wmma<0, true, false><<<dim3(128 / 32, KSPLIT), 128, 0, stream>>>(
        ybuf, HID, wt_zh2, nullptr, zm, nullptr, 128, HID, HID / KSPLIT);
    k_finalize<<<(B_ * (STOCH + DET) + 255) / 256, 256, 0, stream>>>(
        zm, z_h2_b, detf, out, stochf, t);
  }
}
